// selfAttention_72164040508163
// MI455X (gfx1250) — matrix operand order repaired and checked by reading, not compiled, not run on hardware
//
#include <hip/hip_runtime.h>
#include <hip/hip_bf16.h>

typedef __bf16 bf16_t;
typedef __attribute__((ext_vector_type(16))) __bf16 v16bf;
typedef __attribute__((ext_vector_type(8)))  float  v8f;
typedef int v4i __attribute__((ext_vector_type(4)));
typedef int v8i __attribute__((ext_vector_type(8)));

#define N_HEADS 16
#define D_EMBED 1024
#define D_HEAD  64
#define BATCH   4
#define SEQ     2048
#define BS      (BATCH * SEQ)          // 8192 rows
#define SCALE   0.125f                 // 1/sqrt(64)
#define LOG2E   1.44269504088896340736f

// ---------------------------------------------------------------- helpers

__device__ __forceinline__ bf16_t f2bf(float f) {
    // round-to-nearest-even f32 -> bf16
    unsigned u = __float_as_uint(f);
    unsigned r = (u + 0x7FFFu + ((u >> 16) & 1u)) >> 16;
    unsigned short s = (unsigned short)r;
    bf16_t b;
    __builtin_memcpy(&b, &s, 2);
    return b;
}

union Frag {
    v16bf v;
    uint4 q[2];
};

// A-matrix 16x32 bf16 fragment (ISA 7.12.2): lane m=lane&15 holds row m.
// lanes 0-15: K = 0-7 then 16-23 ; lanes 16-31: K = 8-15 then 24-31.
__device__ __forceinline__ v16bf load_a_frag(const bf16_t* base, int stride, int lane) {
    int m  = lane & 15;
    int hi = lane >> 4;
    const bf16_t* row = base + (size_t)m * stride;
    Frag f;
    f.q[0] = *reinterpret_cast<const uint4*>(row + (hi ? 8 : 0));
    f.q[1] = *reinterpret_cast<const uint4*>(row + (hi ? 24 : 16));
    return f.v;
}

// B-matrix 32x16 bf16 fragment; storage is column-major tile: base[n*stride + k].
// lane n=lane&15 holds column n; lanes 0-15: K=0-7 then 16-23 ; lanes 16-31: K=8-15 then 24-31.
__device__ __forceinline__ v16bf load_b_frag(const bf16_t* base, int stride, int lane) {
    int n  = lane & 15;
    int hi = lane >> 4;
    const bf16_t* col = base + (size_t)n * stride;
    Frag f;
    f.q[0] = *reinterpret_cast<const uint4*>(col + (hi ? 8 : 0));
    f.q[1] = *reinterpret_cast<const uint4*>(col + (hi ? 24 : 16));
    return f.v;
}

__device__ __forceinline__ v8f wmma_bf16(v16bf a, v16bf b, v8f c) {
    return __builtin_amdgcn_wmma_f32_16x16x32_bf16(
        /*neg_a=*/false, a, /*neg_b=*/false, b,
        /*c_mod=*/(short)0, c, /*reuse_a=*/false, /*reuse_b=*/false);
}

// gfx1250 async global->LDS copy, tracked by ASYNCcnt (ISA 08_async_tensor §4).
__device__ __forceinline__ void async_b128(unsigned lds_addr, const void* gaddr) {
    asm volatile("global_load_async_to_lds_b128 %0, %1, off"
                 :: "v"(lds_addr), "v"((unsigned long long)gaddr)
                 : "memory");
}
__device__ __forceinline__ void wait_async0() {
    asm volatile("s_wait_asynccnt 0x0" ::: "memory");
}
__device__ __forceinline__ unsigned lds_off(const void* p) {
    return (unsigned)(unsigned long long)p;   // low 32 bits of generic ptr = DS addr
}

// ---------------------------------------------------------------- convert

__global__ void __launch_bounds__(256) cvt_bf16(const float* __restrict__ src,
                                                bf16_t* __restrict__ dst, int n) {
    int i = blockIdx.x * 256 + threadIdx.x;
    if (i < n) dst[i] = f2bf(src[i]);
}

// ---------------------------------------------------------------- qkv GEMM
// C[8192,3072] = Xb[8192,1024] * Wb[3072,1024]^T + bias, scattered into
// head-major bf16 K/Q/V buffers [B,H,S,Dh].  chunks: 0=K, 1=Q, 2=V.
// Block tile 128(M) x 64(N), 8 waves each 32x32, K-step 32,
// double-buffered async global->LDS pipeline.

__global__ void __launch_bounds__(256) qkv_gemm(const bf16_t* __restrict__ Xb,
                                                const bf16_t* __restrict__ Wb,
                                                const float* __restrict__ bias,
                                                bf16_t* __restrict__ Kbuf,
                                                bf16_t* __restrict__ Qbuf,
                                                bf16_t* __restrict__ Vbuf) {
    __shared__ __align__(16) bf16_t As[2][128][32];
    __shared__ __align__(16) bf16_t Bs[2][64][32];

    const int t    = threadIdx.x;
    const int lane = t & 31;
    const int wave = t >> 5;
    const int wm   = wave >> 1;     // 0..3 -> M sub-tile
    const int wn   = wave & 1;      // 0..1 -> N sub-tile
    const int mBase = blockIdx.x * 128;
    const int nBase = blockIdx.y * 64;

    auto issue_tile = [&](int buf, int k0) {
        #pragma unroll
        for (int i = 0; i < 2; ++i) {
            int u = t * 2 + i;
            int r = u >> 2, c = (u & 3) * 8;
            async_b128(lds_off(&As[buf][r][c]),
                       Xb + (size_t)(mBase + r) * D_EMBED + k0 + c);
        }
        int r = t >> 2, c = (t & 3) * 8;
        async_b128(lds_off(&Bs[buf][r][c]),
                   Wb + (size_t)(nBase + r) * D_EMBED + k0 + c);
    };

    v8f acc00 = {}, acc01 = {}, acc10 = {}, acc11 = {};

    issue_tile(0, 0);
    for (int k0 = 0; k0 < D_EMBED; k0 += 32) {
        const int buf = (k0 >> 5) & 1;
        wait_async0();          // my piece of tile k0 landed in LDS
        __syncthreads();        // everyone's piece landed; prev reads of buf^1 done
        if (k0 + 32 < D_EMBED) {
            issue_tile(buf ^ 1, k0 + 32);
            __builtin_prefetch(Xb + (size_t)(mBase + (t >> 1)) * D_EMBED + k0 + 64, 0, 3);
        }

        v16bf a0 = load_a_frag(&As[buf][wm * 32 + 0][0], 32, lane);
        v16bf a1 = load_a_frag(&As[buf][wm * 32 + 16][0], 32, lane);
        v16bf b0 = load_b_frag(&Bs[buf][wn * 32 + 0][0], 32, lane);
        v16bf b1 = load_b_frag(&Bs[buf][wn * 32 + 16][0], 32, lane);
        acc00 = wmma_bf16(a0, b0, acc00);
        acc01 = wmma_bf16(a0, b1, acc01);
        acc10 = wmma_bf16(a1, b0, acc10);
        acc11 = wmma_bf16(a1, b1, acc11);
    }

    // epilogue: bias + scatter into head-major bf16 buffer
    const int chunk = nBase >> 10;                       // uniform per block
    bf16_t* dst = (chunk == 0) ? Kbuf : (chunk == 1) ? Qbuf : Vbuf;
    const int h      = (nBase & 1023) >> 6;              // uniform per block
    const int half   = lane >> 4;
    const int lane15 = lane & 15;

    #pragma unroll
    for (int mi = 0; mi < 2; ++mi) {
        #pragma unroll
        for (int ni = 0; ni < 2; ++ni) {
            const v8f& a = mi ? (ni ? acc11 : acc10) : (ni ? acc01 : acc00);
            #pragma unroll
            for (int j = 0; j < 8; ++j) {
                int gm = mBase + wm * 32 + mi * 16 + j + 8 * half;
                int gn = nBase + wn * 32 + ni * 16 + lane15;
                float val = a[j] + bias[gn];
                int b = gm >> 11, s = gm & 2047;
                int dh = gn & 63;
                size_t idx = (((size_t)(b * N_HEADS + h)) * SEQ + s) * D_HEAD + dh;
                dst[idx] = f2bf(val);
            }
        }
    }
}

// ---------------------------------------------------------------- attention
// One workgroup = one (b,h) x 128 query rows; each of 8 waves owns 16 rows.
// Flash-attention over key blocks of 32. grid = (SEQ/128, B*H).
// K tile (contiguous 4KB) staged by wave 0 via the Tensor Data Mover;
// V tile staged transposed through VGPRs.

__global__ void __launch_bounds__(256) attention(const bf16_t* __restrict__ Qbuf,
                                                 const bf16_t* __restrict__ Kbuf,
                                                 const bf16_t* __restrict__ Vbuf,
                                                 bf16_t* __restrict__ Ab) {
    __shared__ __align__(16) bf16_t Ks[32][64];       // [key][dh]
    __shared__ __align__(16) bf16_t Vs[64][32];       // transposed: [dh][key]
    __shared__ __align__(16) bf16_t Ps[8][16][32];    // per-wave P strip

    const int t      = threadIdx.x;
    const int lane   = t & 31;
    const int wave   = t >> 5;
    const int half   = lane >> 4;
    const int lane15 = lane & 15;
    const int bh     = blockIdx.y;                    // b*16 + h
    const int qRow0  = blockIdx.x * 128 + wave * 16;

    const size_t headBase = (size_t)bh * SEQ * D_HEAD;

    // resident Q fragments (rows qRow0..+15, dh chunks 0-31 / 32-63)
    const bf16_t* qptr = Qbuf + headBase + (size_t)qRow0 * D_HEAD;
    v16bf qA0 = load_a_frag(qptr + 0, D_HEAD, lane);
    v16bf qA1 = load_a_frag(qptr + 32, D_HEAD, lane);

    v8f o0 = {}, o1 = {}, o2 = {}, o3 = {};
    float m_run[8], l_run[8];
    #pragma unroll
    for (int j = 0; j < 8; ++j) { m_run[j] = -1e30f; l_run[j] = 0.0f; }

    const unsigned ldsKs = lds_off(&Ks[0][0]);

    for (int kb0 = 0; kb0 < SEQ; kb0 += 32) {
        __syncthreads();
        // ---- K tile: one TDM transfer of the contiguous 32x64 bf16 block
        if (wave == 0) {
            unsigned long long ga =
                (unsigned long long)(Kbuf + headBase + (size_t)kb0 * D_HEAD);
            v4i g0;
            g0[0] = 1;                                     // count=1, flags=0
            g0[1] = (int)ldsKs;                            // lds_addr
            g0[2] = (int)(unsigned)ga;                     // global_addr[31:0]
            g0[3] = (int)((unsigned)((ga >> 32) & 0x1FFFFFFull) | 0x80000000u); // [56:32] | type=2
            v8i g1;
            g1[0] = 0x00010000;       // workgroup_mask=0, data_size=1 (2 bytes)
            g1[1] = 0;                // atomic barrier addr = 0
            g1[2] = 0x00010002;       // tensor_dim0[31:16]=2 (131072), tensor_dim1[15:0]=1
            g1[3] = 0x08000000;       // tile_dim0 = 2048 elements
            g1[4] = 0x00000001;       // tile_dim1 = 1, tile_dim2 = 0
            g1[5] = 0x00020000;       // tensor_dim0_stride = 131072
            g1[6] = 0;
            g1[7] = 0;
            asm volatile("tensor_load_to_lds %0, %1" :: "s"(g0), "s"(g1) : "memory");
            __builtin_amdgcn_s_wait_tensorcnt(0);
            asm volatile("" ::: "memory");
        }
        // ---- V tile: load rows, store transposed into LDS
        {
            int key = t >> 3, c = (t & 7) * 8;
            uint4 vv =
                *reinterpret_cast<const uint4*>(Vbuf + headBase + (size_t)(kb0 + key) * D_HEAD + c);
            const bf16_t* ve = reinterpret_cast<const bf16_t*>(&vv);
            #pragma unroll
            for (int i = 0; i < 8; ++i) Vs[c + i][key] = ve[i];
        }
        __syncthreads();

        // S = Q @ K^T  (16 x 32 keys per wave, fp32)
        v8f st0 = {}, st1 = {};
        st0 = wmma_bf16(qA0, load_b_frag(&Ks[0][0],  D_HEAD, lane), st0);
        st0 = wmma_bf16(qA1, load_b_frag(&Ks[0][32], D_HEAD, lane), st0);
        st1 = wmma_bf16(qA0, load_b_frag(&Ks[16][0],  D_HEAD, lane), st1);
        st1 = wmma_bf16(qA1, load_b_frag(&Ks[16][32], D_HEAD, lane), st1);

        // online softmax per row (rows live in 16-lane groups; xor masks <16 stay in-group)
        #pragma unroll
        for (int j = 0; j < 8; ++j) {
            float t0 = st0[j] * SCALE;
            float t1 = st1[j] * SCALE;
            float mx = fmaxf(t0, t1);
            mx = fmaxf(mx, __shfl_xor(mx, 1));
            mx = fmaxf(mx, __shfl_xor(mx, 2));
            mx = fmaxf(mx, __shfl_xor(mx, 4));
            mx = fmaxf(mx, __shfl_xor(mx, 8));
            float mnew = fmaxf(m_run[j], mx);
            float corr = exp2f((m_run[j] - mnew) * LOG2E);
            float p0 = exp2f((t0 - mnew) * LOG2E);
            float p1 = exp2f((t1 - mnew) * LOG2E);
            float rs = p0 + p1;
            rs += __shfl_xor(rs, 1);
            rs += __shfl_xor(rs, 2);
            rs += __shfl_xor(rs, 4);
            rs += __shfl_xor(rs, 8);
            l_run[j] = l_run[j] * corr + rs;
            m_run[j] = mnew;
            o0[j] *= corr; o1[j] *= corr; o2[j] *= corr; o3[j] *= corr;
            int r = j + 8 * half;
            Ps[wave][r][lane15]      = f2bf(p0);
            Ps[wave][r][lane15 + 16] = f2bf(p1);
        }

        // O += P @ V  (A = P 16x32, B = V chunks from transposed tile)
        v16bf pA = load_a_frag(&Ps[wave][0][0], 32, lane);
        o0 = wmma_bf16(pA, load_b_frag(&Vs[0][0],  32, lane), o0);
        o1 = wmma_bf16(pA, load_b_frag(&Vs[16][0], 32, lane), o1);
        o2 = wmma_bf16(pA, load_b_frag(&Vs[32][0], 32, lane), o2);
        o3 = wmma_bf16(pA, load_b_frag(&Vs[48][0], 32, lane), o3);
    }

    // finalize: divide by l, write bf16 attention output [B,S,D] (d = h*64+dh)
    const int b = bh >> 4, h = bh & 15;
    #pragma unroll
    for (int j = 0; j < 8; ++j) {
        float inv = 1.0f / l_run[j];
        int r = j + 8 * half;
        size_t grow = (size_t)b * SEQ + qRow0 + r;
        size_t base = grow * D_EMBED + h * D_HEAD + lane15;
        Ab[base + 0]  = f2bf(o0[j] * inv);
        Ab[base + 16] = f2bf(o1[j] * inv);
        Ab[base + 32] = f2bf(o2[j] * inv);
        Ab[base + 48] = f2bf(o3[j] * inv);
    }
}

// ---------------------------------------------------------------- out GEMM
// out[8192,1024] = Ab[8192,1024] * Wout[1024,1024]^T + bias (fp32 out)

__global__ void __launch_bounds__(256) out_gemm(const bf16_t* __restrict__ Ab,
                                                const bf16_t* __restrict__ Wb,
                                                const float* __restrict__ bias,
                                                float* __restrict__ out) {
    __shared__ __align__(16) bf16_t As[2][128][32];
    __shared__ __align__(16) bf16_t Bs[2][64][32];

    const int t    = threadIdx.x;
    const int lane = t & 31;
    const int wave = t >> 5;
    const int wm   = wave >> 1;
    const int wn   = wave & 1;
    const int mBase = blockIdx.x * 128;
    const int nBase = blockIdx.y * 64;

    auto issue_tile = [&](int buf, int k0) {
        #pragma unroll
        for (int i = 0; i < 2; ++i) {
            int u = t * 2 + i;
            int r = u >> 2, c = (u & 3) * 8;
            async_b128(lds_off(&As[buf][r][c]),
                       Ab + (size_t)(mBase + r) * D_EMBED + k0 + c);
        }
        int r = t >> 2, c = (t & 3) * 8;
        async_b128(lds_off(&Bs[buf][r][c]),
                   Wb + (size_t)(nBase + r) * D_EMBED + k0 + c);
    };

    v8f acc00 = {}, acc01 = {}, acc10 = {}, acc11 = {};

    issue_tile(0, 0);
    for (int k0 = 0; k0 < D_EMBED; k0 += 32) {
        const int buf = (k0 >> 5) & 1;
        wait_async0();
        __syncthreads();
        if (k0 + 32 < D_EMBED) {
            issue_tile(buf ^ 1, k0 + 32);
            __builtin_prefetch(Ab + (size_t)(mBase + (t >> 1)) * D_EMBED + k0 + 64, 0, 3);
        }

        v16bf a0 = load_a_frag(&As[buf][wm * 32 + 0][0], 32, lane);
        v16bf a1 = load_a_frag(&As[buf][wm * 32 + 16][0], 32, lane);
        v16bf b0 = load_b_frag(&Bs[buf][wn * 32 + 0][0], 32, lane);
        v16bf b1 = load_b_frag(&Bs[buf][wn * 32 + 16][0], 32, lane);
        acc00 = wmma_bf16(a0, b0, acc00);
        acc01 = wmma_bf16(a0, b1, acc01);
        acc10 = wmma_bf16(a1, b0, acc10);
        acc11 = wmma_bf16(a1, b1, acc11);
    }

    const int half = lane >> 4, lane15 = lane & 15;
    #pragma unroll
    for (int mi = 0; mi < 2; ++mi) {
        #pragma unroll
        for (int ni = 0; ni < 2; ++ni) {
            const v8f& a = mi ? (ni ? acc11 : acc10) : (ni ? acc01 : acc00);
            #pragma unroll
            for (int j = 0; j < 8; ++j) {
                int gm = mBase + wm * 32 + mi * 16 + j + 8 * half;
                int gn = nBase + wn * 32 + ni * 16 + lane15;
                out[(size_t)gm * D_EMBED + gn] = a[j] + bias[gn];
            }
        }
    }
}

// ---------------------------------------------------------------- launch

extern "C" void kernel_launch(void* const* d_in, const int* in_sizes, int n_in,
                              void* d_out, int out_size, void* d_ws, size_t ws_size,
                              hipStream_t stream) {
    (void)in_sizes; (void)n_in; (void)out_size; (void)ws_size;
    const float* x     = (const float*)d_in[0];
    const float* w_in  = (const float*)d_in[1];
    const float* b_in  = (const float*)d_in[2];
    const float* w_out = (const float*)d_in[3];
    const float* b_out = (const float*)d_in[4];

    char* ws = (char*)d_ws;
    bf16_t* Xb   = (bf16_t*)(ws + (size_t)0);
    bf16_t* Wqkv = (bf16_t*)(ws + ((size_t)16 << 20));
    bf16_t* Wout = (bf16_t*)(ws + ((size_t)22 << 20));
    bf16_t* Kb   = (bf16_t*)(ws + ((size_t)24 << 20));
    bf16_t* Qb   = (bf16_t*)(ws + ((size_t)40 << 20));
    bf16_t* Vb   = (bf16_t*)(ws + ((size_t)56 << 20));
    bf16_t* Ab   = (bf16_t*)(ws + ((size_t)72 << 20));

    const int nX  = BS * D_EMBED;          // 8388608
    const int nW1 = 3 * D_EMBED * D_EMBED; // 3145728
    const int nW2 = D_EMBED * D_EMBED;     // 1048576

    cvt_bf16<<<(nX  + 255) / 256, 256, 0, stream>>>(x,     Xb,   nX);
    cvt_bf16<<<(nW1 + 255) / 256, 256, 0, stream>>>(w_in,  Wqkv, nW1);
    cvt_bf16<<<(nW2 + 255) / 256, 256, 0, stream>>>(w_out, Wout, nW2);

    qkv_gemm<<<dim3(BS / 128, (3 * D_EMBED) / 64), 256, 0, stream>>>(
        Xb, Wqkv, b_in, Kb, Qb, Vb);

    attention<<<dim3(SEQ / 128, BATCH * N_HEADS), 256, 0, stream>>>(Qb, Kb, Vb, Ab);

    out_gemm<<<dim3(BS / 128, D_EMBED / 64), 256, 0, stream>>>(Ab, Wout, b_out,
                                                               (float*)d_out);
}
